// MultiInstanceNNEmbedding_3341484556547
// MI455X (gfx1250) — matrix operand order repaired and checked by reading, not compiled, not run on hardware
//
#include <hip/hip_runtime.h>
#include <hip/hip_bf16.h>

typedef __attribute__((ext_vector_type(16))) _Float16 v16h;
typedef __attribute__((ext_vector_type(8)))  _Float16 v8h;
typedef __attribute__((ext_vector_type(8)))  float    v8f;
typedef __attribute__((ext_vector_type(4)))  float    v4f;

#define NREADS 1048576
#define GSITES 32768
#define TILES_PER_WAVE 8      // 16-read tiles processed per wave in kernel 1
#define GROUPS_PER_BLOCK 4    // 16-site groups processed per block in kernel 2

static __device__ __forceinline__ v8f wmma_f16(v16h a, v16h b, v8f c) {
  // 8 args: (neg_a, A, neg_b, B, c_mod, C, reuse_a, reuse_b)
  return __builtin_amdgcn_wmma_f32_16x16x32_f16(false, a, false, b, (short)0, c, false, false);
}

// ---------------------------------------------------------------------------
// Kernel 1: per-read MLP  h = relu(relu([x|emb[kmer]] W1 + b1) W2 + b2)
// 8 tiles of 16 reads per wave, 4 waves per block. K padded 18->32,
// H 150->160, P 8->16. W1/W2 staged once per block in LDS, pre-swizzled into
// the exact WMMA B-fragment layout (one 32B contiguous ds read per lane).
// ---------------------------------------------------------------------------
__global__ __launch_bounds__(128) void mlp_read_kernel(
    const float* __restrict__ x, const int* __restrict__ kmer,
    const float* __restrict__ emb,
    const float* __restrict__ W1, const float* __restrict__ bias1,
    const float* __restrict__ W2, const float* __restrict__ bias2,
    float* __restrict__ hbuf)
{
  __shared__ __align__(64) _Float16 fB1[10][32][16];   // W1 B-fragments (10 col tiles)
  __shared__ __align__(64) _Float16 fB2[5][32][16];    // W2 B-fragments (5 K tiles)
  __shared__ __align__(64) _Float16 tr[4][16][168];    // per-wave C->A transpose buffer

  const int t = threadIdx.x;
  // B layout: lane holds col N=lane&15, half i -> K = (i&7) + (lane>=16 ? 8 : 0) + (i>=8 ? 16 : 0)
  for (int f = t; f < 10*32*16; f += 128) {
    const int i = f & 15, lane = (f >> 4) & 31, c = f >> 9;
    const int k = (i & 7) + ((lane & 16) ? 8 : 0) + ((i & 8) ? 16 : 0);
    const int col = 16*c + (lane & 15);
    fB1[c][lane][i] = (_Float16)((k < 18 && col < 150) ? W1[k*150 + col] : 0.0f);
  }
  for (int f = t; f < 5*32*16; f += 128) {
    const int i = f & 15, lane = (f >> 4) & 31, kt = f >> 9;
    const int k = 32*kt + (i & 7) + ((lane & 16) ? 8 : 0) + ((i & 8) ? 16 : 0);
    const int col = lane & 15;
    fB2[kt][lane][i] = (_Float16)((k < 150 && col < 8) ? W2[k*8 + col] : 0.0f);
  }
  __syncthreads();

  const int wave = t >> 5;
  const int lane = t & 31;
  const int lrow = lane & 15;   // A: row M   / C: col N
  const int hi   = lane >> 4;   // lane-half selector

  // Layer-2 B fragments are loop-invariant: keep them in registers.
  v16h bf2[5];
  #pragma unroll
  for (int kt = 0; kt < 5; kt++) bf2[kt] = *(const v16h*)&fB2[kt][lane][0];

  const float b2v = (lrow < 8) ? bias2[lrow] : 0.0f;
  const long tile0 = ((long)blockIdx.x * 4 + wave) * TILES_PER_WAVE;

  for (int tt = 0; tt < TILES_PER_WAVE; ++tt) {
    const long tile   = tile0 + tt;
    const long read0  = tile * 16;
    const long myread = read0 + lrow;

    // Prefetch next tile's covariate rows (global_prefetch_b8).
    if (tt + 1 < TILES_PER_WAVE)
      __builtin_prefetch(x + (myread + 16) * 16, 0, 1);

    // Build layer-1 A fragment (16x32 f16): K = (i%8) + hi*8 + (i/8)*16
    v16h a1;
    {
      const float* xr = x + myread*16 + hi*8;         // K 0..7 (lo) / 8..15 (hi)
      v4f x0 = *(const v4f*)(xr);
      v4f x1 = *(const v4f*)(xr + 4);
      #pragma unroll
      for (int i = 0; i < 4; i++) { a1[i] = (_Float16)x0[i]; a1[4+i] = (_Float16)x1[i]; }
      float e0 = 0.0f, e1 = 0.0f;
      if (hi == 0) {                                  // K=16,17 live in lanes 0-15 only
        const int km = kmer[myread];
        e0 = emb[km*2]; e1 = emb[km*2 + 1];
      }
      a1[8] = (_Float16)e0; a1[9] = (_Float16)e1;
      #pragma unroll
      for (int i = 10; i < 16; i++) a1[i] = (_Float16)0.0f;
    }

    // Layer 1: 10 col tiles; bias preloaded into C; relu; scatter to transpose buf
    #pragma unroll
    for (int c = 0; c < 10; c++) {
      const int col = 16*c + lrow;
      const float bv = (col < 150) ? bias1[col] : 0.0f;
      v8f acc = { bv,bv,bv,bv,bv,bv,bv,bv };
      const v16h bf = *(const v16h*)&fB1[c][lane][0];
      acc = wmma_f16(a1, bf, acc);
      #pragma unroll
      for (int r = 0; r < 8; r++) {
        const float v = acc[r];                       // C elem: M=r+hi*8, N=col
        tr[wave][r + hi*8][col] = (_Float16)(v > 0.0f ? v : 0.0f);
      }
    }

    // Per-wave producer->consumer through LDS: DS ops are in-order per wave;
    // the explicit wait also stops the compiler from hoisting the reloads.
    asm volatile("s_wait_dscnt 0" ::: "memory");

    // Layer 2: K = 160 -> 5 K-tiles of 32, accumulate one 16x16 tile
    v8f acc2 = { b2v,b2v,b2v,b2v,b2v,b2v,b2v,b2v };
    #pragma unroll
    for (int kt = 0; kt < 5; kt++) {
      const int kb = 32*kt + hi*8;
      const v8h lo = *(const v8h*)&tr[wave][lrow][kb];       // K kb..kb+7
      const v8h hh = *(const v8h*)&tr[wave][lrow][kb + 16];  // K kb+16..kb+23
      v16h a2;
      #pragma unroll
      for (int i = 0; i < 8; i++) { a2[i] = lo[i]; a2[8+i] = hh[i]; }
      acc2 = wmma_f16(a2, bf2[kt], acc2);
    }
    if (lrow < 8) {                                   // cols 8..15 are padding
      #pragma unroll
      for (int r = 0; r < 8; r++) {
        const float v = acc2[r];
        hbuf[(read0 + r + hi*8)*8 + lrow] = (v > 0.0f ? v : 0.0f);
      }
    }
  }
}

// ---------------------------------------------------------------------------
// Kernel 2: per-site stats (mean/var/min/median/max over 32 reads) + site MLP
// 4 groups of 16 sites per block (128 threads = one (site,feature) each).
// ---------------------------------------------------------------------------
__global__ __launch_bounds__(128) void site_kernel(
    const float* __restrict__ hbuf, const int* __restrict__ indices,
    const float* __restrict__ W3, const float* __restrict__ bias3,
    const float* __restrict__ W4, const float* __restrict__ bias4,
    float* __restrict__ out)
{
  __shared__ __align__(64) _Float16 fB3[2][10][32][16]; // W3 B-fragments
  __shared__ __align__(64) _Float16 agg[16][72];        // [site][40 agg, K-pad to 64, stride 72]
  __shared__ __align__(64) float    zb[16][160];        // layer-3 activations

  const int t = threadIdx.x;
  for (int f = t; f < 2*10*32*16; f += 128) {
    const int i = f & 15, lane = (f >> 4) & 31, ft = f >> 9;
    const int kt = ft / 10, c = ft % 10;
    const int k = 32*kt + (i & 7) + ((lane & 16) ? 8 : 0) + ((i & 8) ? 16 : 0);
    const int col = 16*c + (lane & 15);
    fB3[kt][c][lane][i] = (_Float16)((k < 40 && col < 150) ? W3[k*150 + col] : 0.0f);
  }
  for (int f = t; f < 16*72; f += 128) agg[f / 72][f % 72] = (_Float16)0.0f;
  __syncthreads();

  const int lane = t & 31, wave = t >> 5;
  const int lrow = lane & 15, hi = lane >> 4;
  const int sl = t >> 3, feat = t & 7;

  for (int g = 0; g < GROUPS_PER_BLOCK; ++g) {
    const long site0 = (long)blockIdx.x * (16*GROUPS_PER_BLOCK) + g*16;

    // ---- Stats: one (site, feature) per thread, register bitonic sort
    {
      const long site = site0 + sl;
      const int* idxp = indices + site * 32;
      float v[32];
      #pragma unroll
      for (int m = 0; m < 32; m++) v[m] = hbuf[(long)idxp[m]*8 + feat];
      float sum = 0.0f, sq = 0.0f;
      #pragma unroll
      for (int m = 0; m < 32; m++) { sum += v[m]; sq += v[m]*v[m]; }
      const float mean = sum * (1.0f/32.0f);
      const float var  = (sq - sum*mean) * (1.0f/31.0f);   // unbiased (ddof=1)
      // Fully-unrolled bitonic sort, static indices -> stays in VGPRs
      #pragma unroll
      for (int k = 2; k <= 32; k <<= 1) {
        #pragma unroll
        for (int j = k >> 1; j > 0; j >>= 1) {
          #pragma unroll
          for (int i = 0; i < 32; i++) {
            const int ixj = i ^ j;
            if (ixj > i) {
              const float a = v[i], b = v[ixj];
              const float lo = fminf(a, b), hi2 = fmaxf(a, b);
              const bool up = (i & k) == 0;
              v[i]   = up ? lo  : hi2;
              v[ixj] = up ? hi2 : lo;
            }
          }
        }
      }
      agg[sl][ 0 + feat] = (_Float16)mean;
      agg[sl][ 8 + feat] = (_Float16)var;
      agg[sl][16 + feat] = (_Float16)v[0];    // min
      agg[sl][24 + feat] = (_Float16)v[15];   // lower median, (M-1)//2
      agg[sl][32 + feat] = (_Float16)v[31];   // max
    }
    __syncthreads();

    // ---- Layer 3: [16,40(->64)] @ W3 -> [16,160], col tiles split across waves
    for (int c = wave; c < 10; c += 4) {
      const int col = 16*c + lrow;
      const float bv = (col < 150) ? bias3[col] : 0.0f;
      v8f acc = { bv,bv,bv,bv,bv,bv,bv,bv };
      #pragma unroll
      for (int kt = 0; kt < 2; kt++) {
        const int kb = 32*kt + hi*8;
        const v8h lo = *(const v8h*)&agg[lrow][kb];
        const v8h hh = *(const v8h*)&agg[lrow][kb + 16];
        v16h a;
        #pragma unroll
        for (int i = 0; i < 8; i++) { a[i] = lo[i]; a[8+i] = hh[i]; }
        const v16h bf = *(const v16h*)&fB3[kt][c][lane][0];
        acc = wmma_f16(a, bf, acc);
      }
      #pragma unroll
      for (int r = 0; r < 8; r++) {
        const float v = acc[r];
        zb[r + hi*8][col] = (v > 0.0f ? v : 0.0f);
      }
    }
    __syncthreads();

    // ---- Layer 4 + sigmoid: tiny dot product per site
    if (t < 16) {
      float s = bias4[0];
      for (int k = 0; k < 150; k++) s += zb[t][k] * W4[k];
      out[site0 + t] = 1.0f / (1.0f + __expf(-s));
    }
    __syncthreads();   // protect agg/zb before next group overwrites them
  }
}

extern "C" void kernel_launch(void* const* d_in, const int* in_sizes, int n_in,
                              void* d_out, int out_size, void* d_ws, size_t ws_size,
                              hipStream_t stream) {
  const float* x     = (const float*)d_in[0];
  const int*   kmer  = (const int*)d_in[1];
  const int*   idx   = (const int*)d_in[2];
  const float* emb   = (const float*)d_in[3];
  const float* W1    = (const float*)d_in[4];
  const float* bias1 = (const float*)d_in[5];
  const float* W2    = (const float*)d_in[6];
  const float* bias2 = (const float*)d_in[7];
  const float* W3    = (const float*)d_in[8];
  const float* bias3 = (const float*)d_in[9];
  const float* W4    = (const float*)d_in[10];
  const float* bias4 = (const float*)d_in[11];
  float* out  = (float*)d_out;
  float* hbuf = (float*)d_ws;   // N*8 floats = 32 MB intermediate h

  (void)in_sizes; (void)n_in; (void)out_size; (void)ws_size;

  const dim3 blk(128);
  const dim3 g1((NREADS / 16) / (4 * TILES_PER_WAVE));  // 2048 blocks
  mlp_read_kernel<<<g1, blk, 0, stream>>>(x, kmer, emb, W1, bias1, W2, bias2, hbuf);

  const dim3 g2(GSITES / (16 * GROUPS_PER_BLOCK));      // 512 blocks
  site_kernel<<<g2, blk, 0, stream>>>(hbuf, idx, W3, bias3, W4, bias4, out);
}
